// MyModel_69277822485198
// MI455X (gfx1250) — compile-verified, matrix operand order checked by reading, not run on hardware
//
#include <hip/hip_runtime.h>

// ---------------------------------------------------------------------------
// Types
// ---------------------------------------------------------------------------
typedef __bf16 v8bf  __attribute__((ext_vector_type(8)));
typedef __bf16 v16bf __attribute__((ext_vector_type(16)));
typedef float  v8f   __attribute__((ext_vector_type(8)));
typedef unsigned int u32x4 __attribute__((ext_vector_type(4)));
typedef int  i32x4 __attribute__((ext_vector_type(4)));
typedef int  i32x8 __attribute__((ext_vector_type(8)));

union ABFrag { v16bf v; v8bf h[2]; };

__device__ __forceinline__ __bf16 f2bf(float f) {
    union { float f; unsigned u; } v; v.f = f;
    unsigned r = (v.u + 0x7FFFu + ((v.u >> 16) & 1u)) >> 16;
    unsigned short s = (unsigned short)r;
    return __builtin_bit_cast(__bf16, s);
}

static constexpr int Himg = 224, Wimg = 224, HW = Himg * Wimg;
static constexpr int NB = 214;

// ---------------------------------------------------------------------------
// TDM: async 2D tile load (rows x row_elems bf16, row stride in elements)
// into LDS at lds_off. Tracked by TENSORcnt.
// ---------------------------------------------------------------------------
__device__ __forceinline__ void tdm_load_2d(unsigned lds_off, const void* gptr,
                                            unsigned row_elems, unsigned rows,
                                            unsigned row_stride_elems) {
    unsigned long long ga = (unsigned long long)(size_t)gptr;
    u32x4 g0;
    g0[0] = 1u;                                              // count=1, user mode
    g0[1] = lds_off;                                         // LDS byte address
    g0[2] = (unsigned)ga;                                    // global addr lo
    g0[3] = (unsigned)((ga >> 32) & 0x01FFFFFFu) | (2u << 30); // addr hi | type=2
    i32x8 g1;
    g1[0] = (int)(1u << 16);                                 // data_size=1 (2B)
    g1[1] = (int)((row_elems & 0xFFFFu) << 16);              // tensor_dim0 lo16
    g1[2] = (int)((row_elems >> 16) | ((rows & 0xFFFFu) << 16)); // dim0 hi | dim1 lo
    g1[3] = (int)((rows >> 16) | ((row_elems & 0xFFFFu) << 16)); // dim1 hi | tile_dim0
    g1[4] = (int)(rows & 0xFFFFu);                           // tile_dim1 | tile_dim2=0
    g1[5] = (int)row_stride_elems;                           // tensor_dim0_stride lo32
    g1[6] = 0;                                               // stride hi | dim1_stride
    g1[7] = 0;
    i32x4 z4 = {};
#if __clang_major__ >= 23
    i32x8 z8 = {};
    __builtin_amdgcn_tensor_load_to_lds(g0, g1, z4, z4, z8, 0);
#else
    __builtin_amdgcn_tensor_load_to_lds(g0, g1, z4, z4, 0);
#endif
}

// ---------------------------------------------------------------------------
// zero workspace region (counts | pooled | val)
// ---------------------------------------------------------------------------
__global__ void zero_kernel(float* __restrict__ z, int n) {
    int i = blockIdx.x * 256 + threadIdx.x;
    if (i < n) z[i] = 0.0f;
}

// ---------------------------------------------------------------------------
// histogram of bin indices ("+1" applied at consume time)
// ---------------------------------------------------------------------------
__global__ void hist_kernel(const int* __restrict__ p, float* __restrict__ counts) {
    int i = blockIdx.x * 256 + threadIdx.x;
    if (i < HW) atomicAdd(&counts[p[i]], 1.0f);
}

// ---------------------------------------------------------------------------
// pack OIHW fp32 weights -> bf16 [co][tap][ci], co zero-padded to COUTPAD
// ---------------------------------------------------------------------------
__global__ void pack_w_kernel(const float* __restrict__ w, __bf16* __restrict__ out,
                              int COUT, int COUTPAD, int CIN) {
    int idx = blockIdx.x * 256 + threadIdx.x;
    int KP = CIN * 9;
    if (idx >= COUTPAD * KP) return;
    int co = idx / KP, k = idx % KP;
    int t = k / CIN, ci = k % CIN;
    float v = (co < COUT) ? w[(size_t)(co * CIN + ci) * 9 + t] : 0.0f;
    out[idx] = f2bf(v);
}

// ---------------------------------------------------------------------------
// conv1: 3 -> 64 direct (K=27, 1.4 GFLOP). Writes NHWC bf16.
// ---------------------------------------------------------------------------
__global__ __launch_bounds__(256) void conv1_kernel(
    const float* __restrict__ x, const float* __restrict__ w1,
    const float* __restrict__ b1, __bf16* __restrict__ h1) {
    __shared__ float wsh[16 * 27];
    const int tid = threadIdx.x;
    const int co0 = blockIdx.y * 16;
    for (int i = tid; i < 16 * 27; i += 256)
        wsh[i] = w1[(size_t)(co0 + i / 27) * 27 + (i % 27)];
    __syncthreads();

    const int m = blockIdx.x * 256 + tid;
    const int b = blockIdx.z;
    const int y = m / Wimg, xq = m % Wimg;

    float in[27];
#pragma unroll
    for (int ci = 0; ci < 3; ci++)
#pragma unroll
        for (int t = 0; t < 9; t++) {
            int yy = y + t / 3 - 1, xx = xq + t % 3 - 1;
            in[ci * 9 + t] = (yy >= 0 && yy < Himg && xx >= 0 && xx < Wimg)
                ? x[(((size_t)b * 3 + ci) * Himg + yy) * Wimg + xx] : 0.0f;
        }
#pragma unroll
    for (int co = 0; co < 16; co++) {
        float a = b1[co0 + co];
#pragma unroll
        for (int k = 0; k < 27; k++) a = fmaf(in[k], wsh[co * 27 + k], a);
        a = a > 0.0f ? a : 0.0f;
        h1[((size_t)b * HW + m) * 64 + co0 + co] = f2bf(a);
    }
}

// ---------------------------------------------------------------------------
// Implicit-GEMM conv, bf16 WMMA 16x16x32 (f32 accum).
// Block = 8 waves = 128 pixels x 32 channels; each wave: 16px x 32ch
// (two WMMAs per K-step sharing one A fragment).
// Weights streamed per-tap by the Tensor Data Mover into double-buffered LDS.
// ---------------------------------------------------------------------------
template <int CIN, int COUTPAD, bool LAST>
__global__ __launch_bounds__(256) void conv_wmma_kernel(
    const __bf16* __restrict__ hin,   // [B][HW][CIN] NHWC bf16
    const __bf16* __restrict__ wpk,   // [COUTPAD][CIN*9] bf16 packed
    const float*  __restrict__ bias,  // [COUT]
    int COUT,
    __bf16* __restrict__ hout,        // [B][HW][COUTPAD]   (!LAST)
    float*  __restrict__ pooled,      // [B][COUTPAD]       (LAST)
    float*  __restrict__ val)         // [B][HW]            (LAST)
{
    constexpr int KPAD = CIN * 9;
    __shared__ alignas(16) __bf16 wlds[2 * 32 * CIN];   // double-buffered tap tile
    __shared__ float pooled_sh[32];

    const int tid  = threadIdx.x;
    const int wave = tid >> 5;
    const int lane = tid & 31;
    const int b    = blockIdx.z;
    const int n0   = blockIdx.y * 32;                // output-channel tile base

    const unsigned lds_base = (unsigned)(size_t)(void*)&wlds[0];
    const __bf16* wg = wpk + (size_t)n0 * KPAD;      // this block's 32 channels

    // Preload tap 0 via TDM (wave 0 issues; EXEC is ignored for tensor ops).
    if (tid < 32) {
        tdm_load_2d(lds_base, wg, CIN, 32, KPAD);
        __builtin_amdgcn_s_wait_tensorcnt(0);
    }
    if (tid < 32) pooled_sh[tid] = 0.0f;
    __syncthreads();

    const int mw  = blockIdx.x * 128 + wave * 16;    // wave's 16-pixel M tile
    const int row = lane & 15;                       // M-row / N-col index
    const int hi  = lane >> 4;                       // lane half (K interleave)
    const int kk  = hi * 8;                          // lane K base within step
    const int y   = mw / Wimg;
    const int x   = mw % Wimg + row;                 // this lane's pixel x

    const __bf16* hb = hin + (size_t)b * HW * CIN;
    const v8bf zv = {};
    v8f acc0 = {}, acc1 = {};

    for (int c = 0; c < 9; c++) {                    // one chunk == one 3x3 tap
        const __bf16* wb = &wlds[(c & 1) * 32 * CIN];
        const int yy = y + c / 3 - 1;
        const int xx = x + c % 3 - 1;
        const bool ok = (yy >= 0) & (yy < Himg) & (xx >= 0) & (xx < Wimg);
        const __bf16* ap = hb + ((size_t)yy * Wimg + xx) * CIN + kk;

        // Pipeline: kick off next tap's weight tile while computing this one.
        if (c + 1 < 9 && tid < 32)
            tdm_load_2d(lds_base + (unsigned)(((c + 1) & 1) * 32 * CIN * 2),
                        wg + (size_t)(c + 1) * CIN, CIN, 32, KPAD);
        if (ok) __builtin_prefetch(ap + Wimg * CIN, 0, 0);

#pragma unroll
        for (int kc = 0; kc < CIN; kc += 32) {
            ABFrag a, w0, w1;
            a.h[0] = ok ? *(const v8bf*)(ap + kc)      : zv;  // K = kc+kk..+7
            a.h[1] = ok ? *(const v8bf*)(ap + kc + 16) : zv;  // K = kc+kk+16..+23

            const __bf16* wp0 = &wb[row * CIN + kc + kk];
            const __bf16* wp1 = &wb[(16 + row) * CIN + kc + kk];
            w0.h[0] = *(const v8bf*)(wp0);  w0.h[1] = *(const v8bf*)(wp0 + 16);
            w1.h[0] = *(const v8bf*)(wp1);  w1.h[1] = *(const v8bf*)(wp1 + 16);

            acc0 = __builtin_amdgcn_wmma_f32_16x16x32_bf16(
                false, a.v, false, w0.v, (short)0, acc0, false, false);
            acc1 = __builtin_amdgcn_wmma_f32_16x16x32_bf16(
                false, a.v, false, w1.v, (short)0, acc1, false, false);
        }

        if (c + 1 < 9) {
            if (tid < 32) __builtin_amdgcn_s_wait_tensorcnt(0);
            __syncthreads();
        }
    }

    const int ch0 = n0 + row;        // this lane's two output channels
    const int ch1 = n0 + 16 + row;
    const float bs0 = (ch0 < COUT) ? bias[ch0] : 0.0f;
    const float bs1 = (ch1 < COUT) ? bias[ch1] : 0.0f;

    if constexpr (!LAST) {
#pragma unroll
        for (int r = 0; r < 8; r++) {
            const int pix = mw + r + hi * 8;         // D layout: lanes>=16 -> M+8
            float v0 = acc0[r] + bs0; v0 = v0 > 0.0f ? v0 : 0.0f;
            float v1 = acc1[r] + bs1; v1 = v1 > 0.0f ? v1 : 0.0f;
            __bf16* o = hout + ((size_t)b * HW + pix) * COUTPAD + ch0;
            o[0]  = f2bf(v0);
            o[16] = f2bf(v1);
        }
    } else {
        float ps0 = 0.0f, ps1 = 0.0f;
#pragma unroll
        for (int r = 0; r < 8; r++) {
            float v0 = acc0[r] + bs0; v0 = v0 > 0.0f ? v0 : 0.0f;
            float v1 = acc1[r] + bs1; v1 = v1 > 0.0f ? v1 : 0.0f;
            ps0 += v0; ps1 += v1;
            // sum this pixel's 32 channels across the half-wave
            float s = v0 + v1;
            s += __shfl_xor(s, 1, 32);
            s += __shfl_xor(s, 2, 32);
            s += __shfl_xor(s, 4, 32);
            s += __shfl_xor(s, 8, 32);
            if (row == 0) {
                const int pix = mw + r + hi * 8;
                atomicAdd(&val[(size_t)b * HW + pix], s);
            }
        }
        atomicAdd(&pooled_sh[row], ps0);
        atomicAdd(&pooled_sh[16 + row], ps1);
        __syncthreads();
        if (tid < 32)
            atomicAdd(&pooled[(size_t)b * COUTPAD + n0 + tid], pooled_sh[tid]);
    }
}

// ---------------------------------------------------------------------------
// head: out1 = softmax(GAP @ wc^T + bc)
// ---------------------------------------------------------------------------
__global__ void head_kernel(const float* __restrict__ pooled,
                            const float* __restrict__ wc,
                            const float* __restrict__ bc,
                            float* __restrict__ out, int COUTPAD) {
    int t = threadIdx.x;
    if (t < 16) {
        int b = t >> 1, j = t & 1;
        float acc = bc[j];
        const float inv = 1.0f / (float)HW;
        for (int c = 0; c < NB; c++)
            acc = fmaf(pooled[(size_t)b * COUTPAD + c] * inv, wc[j * NB + c], acc);
        float other = __shfl_xor(acc, 1, 32);
        float mx = fmaxf(acc, other);
        float e = __expf(acc - mx), eo = __expf(other - mx);
        out[b * 2 + j] = e / (e + eo);
    }
}

// ---------------------------------------------------------------------------
// result[b,c,j,k] = val[b,j,k] / (1 + counts[c])   (343 MB, float4 streams)
// ---------------------------------------------------------------------------
__global__ __launch_bounds__(256) void result_kernel(
    const float* __restrict__ val, const float* __restrict__ counts,
    float* __restrict__ out) {
    const int c = blockIdx.y, b = blockIdx.z;
    const int i4 = blockIdx.x * 256 + threadIdx.x;   // < 12544
    const float rec = 1.0f / (1.0f + counts[c]);
    const float4* v4 = (const float4*)(val + (size_t)b * HW);
    float4 v = v4[i4];
    v.x *= rec; v.y *= rec; v.z *= rec; v.w *= rec;
    float4* o4 = (float4*)(out + 16 + ((size_t)b * NB + c) * HW);
    o4[i4] = v;
}

// ---------------------------------------------------------------------------
// Workspace layout (bytes): see offsets below. Total ~156.4 MB.
// ---------------------------------------------------------------------------
extern "C" void kernel_launch(void* const* d_in, const int* in_sizes, int n_in,
                              void* d_out, int out_size, void* d_ws, size_t ws_size,
                              hipStream_t stream) {
    const float* x  = (const float*)d_in[0];
    const int*   p  = (const int*)  d_in[1];
    const float* w1 = (const float*)d_in[2];
    const float* b1 = (const float*)d_in[3];
    const float* w2 = (const float*)d_in[4];
    const float* b2 = (const float*)d_in[5];
    const float* w3 = (const float*)d_in[6];
    const float* b3 = (const float*)d_in[7];
    const float* wc = (const float*)d_in[8];
    const float* bc = (const float*)d_in[9];
    float* out = (float*)d_out;

    char* ws = (char*)d_ws;
    float*  counts = (float*)(ws + 0);            //    256 f
    float*  pooled = (float*)(ws + 1024);         //  8*224 f
    float*  val    = (float*)(ws + 8192);         // 8*50176 f
    __bf16* w2p    = (__bf16*)(ws + 1613824);     // 128*576 bf16
    __bf16* w3p    = (__bf16*)(ws + 1761280);     // 224*1152 bf16
    __bf16* h1     = (__bf16*)(ws + 2277376);     // 8*50176*64 bf16
    __bf16* h2     = (__bf16*)(ws + 53657600);    // 8*50176*128 bf16

    zero_kernel<<<(403456 + 255) / 256, 256, 0, stream>>>((float*)ws, 403456);
    hist_kernel<<<196, 256, 0, stream>>>(p, counts);
    pack_w_kernel<<<(128 * 576 + 255) / 256, 256, 0, stream>>>(w2, w2p, 128, 128, 64);
    pack_w_kernel<<<(224 * 1152 + 255) / 256, 256, 0, stream>>>(w3, w3p, 214, 224, 128);

    conv1_kernel<<<dim3(196, 4, 8), 256, 0, stream>>>(x, w1, b1, h1);

    conv_wmma_kernel<64, 128, false><<<dim3(392, 4, 8), 256, 0, stream>>>(
        h1, w2p, b2, 128, h2, nullptr, nullptr);
    conv_wmma_kernel<128, 224, true><<<dim3(392, 7, 8), 256, 0, stream>>>(
        h2, w3p, b3, 214, nullptr, pooled, val);

    head_kernel<<<1, 32, 0, stream>>>(pooled, wc, bc, out, 224);
    result_kernel<<<dim3(49, 214, 8), 256, 0, stream>>>(val, counts, out);
}
